// ConvLSTM_74646531605160
// MI455X (gfx1250) — matrix operand order repaired and checked by reading, not compiled, not run on hardware
//
#include <hip/hip_runtime.h>
#include <hip/hip_bf16.h>

typedef __attribute__((ext_vector_type(16))) __bf16          v16bf;
typedef __attribute__((ext_vector_type(8)))  float           v8f;
typedef __attribute__((ext_vector_type(8)))  unsigned short  v8u;

__device__ __forceinline__ unsigned short f2bf(float f) {
    unsigned u = __float_as_uint(f);
    u += 0x7FFFu + ((u >> 16) & 1u);          // round to nearest even
    return (unsigned short)(u >> 16);
}
__device__ __forceinline__ float bf2f(unsigned short h) {
    return __uint_as_float(((unsigned)h) << 16);
}
// branchless fast sigmoid/tanh: v_exp_f32 + raw v_rcp_f32, no libm branches
__device__ __forceinline__ float fast_sigmoid(float x) {
    return __builtin_amdgcn_rcpf(1.0f + __expf(-x));
}
__device__ __forceinline__ float fast_tanh(float x) {
    // tanh(x) = 1 - 2/(exp(2x)+1); exp overflow/underflow saturates to +/-1
    return 1.0f - 2.0f * __builtin_amdgcn_rcpf(1.0f + __expf(2.0f * x));
}
__device__ __forceinline__ v16bf load32bf(const unsigned short* p0,
                                          const unsigned short* p1) {
    union { v8u u[2]; v16bf b; } U;
    U.u[0] = *(const v8u*)p0;
    U.u[1] = *(const v8u*)p1;
    return U.b;
}

// ---------------------------------------------------------------------------
// Weight pack: W[256][CIN][3][3] f32 -> Ap[m_p][KPAD] bf16, gate-permuted:
//   m_p = hd*4 + g  with orig channel = g*64 + hd   (g: 0=i,1=f,2=o,3=g)
// K index = c*9 + ky*3 + kx (matches flattened last three dims of W).
// ---------------------------------------------------------------------------
template<int KPAD, int KREAL>
__global__ __launch_bounds__(256) void pack_kernel(const float* __restrict__ W,
                                                   unsigned short* __restrict__ Ap) {
    int idx = blockIdx.x * 256 + threadIdx.x;
    if (idx >= 256 * KPAD) return;
    int mp = idx / KPAD;
    int k  = idx - mp * KPAD;
    unsigned short v = 0;
    if (k < KREAL) {
        int hd = mp >> 2, g = mp & 3;
        int m = g * 64 + hd;
        v = f2bf(W[m * KREAL + k]);
    }
    Ap[mp * KPAD + k] = v;
}

__global__ __launch_bounds__(256) void zero_kernel(float* __restrict__ p, int n) {
    int i = blockIdx.x * 256 + threadIdx.x;
    if (i < n) p[i] = 0.0f;
}

// ---------------------------------------------------------------------------
// ConvLSTM cell: gates = conv3x3(concat(inA,inB)) + b ; LSTM pointwise.
// Implicit GEMM: M=256 (permuted gates) x N=32 pixels x K=Cin*9 (bf16 WMMA).
// FIRST: inputs = x_t (f32, 4ch) ++ h0_prev (bf16, 64ch), K=612 pad 640
// else : inputs = h0_next (bf16,64) ++ h1_prev (bf16,64),  K=1152
// Per wave: 32 gate rows x 32 pixels = 4 WMMA tiles; 8 waves cover M=256.
// ---------------------------------------------------------------------------
template<bool FIRST>
__global__ __launch_bounds__(256) void cell_kernel(
    const float* __restrict__ xsrc,           // FIRST only (x_t)
    const unsigned short* __restrict__ inA,   // !FIRST: h0_next
    const unsigned short* __restrict__ inB,   // FIRST: h0_prev ; !FIRST: h1_prev
    const unsigned short* __restrict__ Apack, // packed bf16 weights
    const float* __restrict__ bias,           // 256 (original gate order)
    float* __restrict__ c_state,              // [64][16384] f32, in-place
    unsigned short* __restrict__ h_out)       // [64][16384] bf16 (next buffer)
{
    constexpr int CTOT   = FIRST ? 68 : 128;
    constexpr int KREAL  = CTOT * 9;                  // 612 / 1152
    constexpr int KPAD   = FIRST ? 640 : 1152;
    constexpr int KSTEPS = KPAD / 32;                 // 20 / 36
    constexpr int ROWLEN = KPAD + 8;                  // +16B: stagger LDS banks

    __shared__ unsigned short Bsm[32 * ROWLEN];       // im2col panel, K-major

    const int tid     = threadIdx.x;
    const int wg      = blockIdx.x;
    const int pixbase = wg * 32;                      // 32 pixels, one image row
    const int r0      = pixbase >> 7;
    const int col0    = pixbase & 127;

    // ---- build im2col panel in LDS: Bsm[n][k] = in[c][r0+ky-1][col0+n+kx-1]
    for (int idx = tid; idx < 32 * KPAD; idx += 256) {
        int n = idx / KPAD;
        int k = idx - n * KPAD;
        unsigned short v = 0;
        if (!FIRST || k < KREAL) {
            int c   = k / 9;
            int rem = k - c * 9;
            int ky  = rem / 3;
            int kx  = rem - ky * 3;
            int rr  = r0 + ky - 1;
            int cc  = col0 + n + kx - 1;
            if (rr >= 0 && rr < 128 && cc >= 0 && cc < 128) {
                int sp = rr * 128 + cc;
                if (FIRST) {
                    v = (c < 4) ? f2bf(xsrc[c * 16384 + sp])
                                : inB[(c - 4) * 16384 + sp];
                } else {
                    v = (c < 64) ? inA[c * 16384 + sp]
                                 : inB[(c - 64) * 16384 + sp];
                }
            }
        }
        Bsm[n * ROWLEN + k] = v;
    }
    __syncthreads();

    const int lane = tid & 31;
    const int ln   = lane & 15;
    const int hs   = lane >> 4;       // half-wave select
    const int wv   = tid >> 5;
    const int m0   = wv * 32;         // this wave's 32 packed gate rows

    v8f z = {0.f,0.f,0.f,0.f,0.f,0.f,0.f,0.f};
    v8f acc[2][2];
    acc[0][0] = z; acc[0][1] = z; acc[1][0] = z; acc[1][1] = z;

    // ---- K loop: 4x WMMA 16x16x32 bf16 per step (2 M-tiles x 2 N-tiles)
    for (int ks = 0; ks < KSTEPS; ++ks) {
        const int k0 = ks * 32;
        v16bf afr[2], bfr[2];
#pragma unroll
        for (int mi = 0; mi < 2; ++mi) {
            // A fragment (16M x 32K): per-lane chunks K=base..+7, K=base+16..+23
            const unsigned short* ap =
                Apack + (m0 + mi * 16 + ln) * KPAD + k0 + hs * 8;
            afr[mi] = load32bf(ap, ap + 16);
        }
#pragma unroll
        for (int ni = 0; ni < 2; ++ni) {
            // B fragment (32K x 16N): per-lane chunks K=base..+7, K=base+16..+23
            const unsigned short* bp =
                &Bsm[(ni * 16 + ln) * ROWLEN + k0 + hs * 8];
            bfr[ni] = load32bf(bp, bp + 16);
        }
#pragma unroll
        for (int mi = 0; mi < 2; ++mi)
#pragma unroll
            for (int ni = 0; ni < 2; ++ni)
                acc[mi][ni] = __builtin_amdgcn_wmma_f32_16x16x32_bf16(
                    false, afr[mi], false, bfr[ni], (short)0, acc[mi][ni],
                    false, false);
    }

    // ---- LSTM pointwise: each lane holds two full (i,f,o,g) quadruples/tile
#pragma unroll
    for (int mi = 0; mi < 2; ++mi) {
        const int mbase = m0 + mi * 16 + hs * 8;   // packed row of acc element 0
#pragma unroll
        for (int ni = 0; ni < 2; ++ni) {
            const int pix = pixbase + ni * 16 + ln;
#pragma unroll
            for (int q = 0; q < 2; ++q) {
                const int hd = (mbase >> 2) + q;
                float ci = acc[mi][ni][q * 4 + 0] + bias[hd];
                float cf = acc[mi][ni][q * 4 + 1] + bias[64 + hd];
                float co = acc[mi][ni][q * 4 + 2] + bias[128 + hd];
                float cg = acc[mi][ni][q * 4 + 3] + bias[192 + hd];
                float ig = fast_sigmoid(ci);
                float fg = fast_sigmoid(cf);
                float og = fast_sigmoid(co);
                float gg = fast_tanh(cg);
                float cp = c_state[hd * 16384 + pix];
                float cn = fg * cp + ig * gg;
                c_state[hd * 16384 + pix] = cn;
                h_out[hd * 16384 + pix]   = f2bf(og * fast_tanh(cn));
            }
        }
    }
}

// ---------------------------------------------------------------------------
// Readout: y = conv3x3(h1, Wr) + br, 64ch -> 1ch. M=1: plain VALU kernel.
// ---------------------------------------------------------------------------
__global__ __launch_bounds__(256) void readout_kernel(
    const unsigned short* __restrict__ h1, const float* __restrict__ Wr,
    const float* __restrict__ br, float* __restrict__ y)
{
    __shared__ float wsm[576];
    for (int i = threadIdx.x; i < 576; i += 256) wsm[i] = Wr[i];
    __syncthreads();

    int p   = blockIdx.x * 256 + threadIdx.x;
    int row = p >> 7, col = p & 127;
    float a = br[0];
    for (int c = 0; c < 64; ++c) {
#pragma unroll
        for (int ky = 0; ky < 3; ++ky) {
            int rr = row + ky - 1;
            if (rr < 0 || rr > 127) continue;
#pragma unroll
            for (int kx = 0; kx < 3; ++kx) {
                int cc = col + kx - 1;
                if (cc < 0 || cc > 127) continue;
                a += bf2f(h1[c * 16384 + rr * 128 + cc]) * wsm[(c * 3 + ky) * 3 + kx];
            }
        }
    }
    y[p] = a;
}

// ---------------------------------------------------------------------------
extern "C" void kernel_launch(void* const* d_in, const int* in_sizes, int n_in,
                              void* d_out, int out_size, void* d_ws, size_t ws_size,
                              hipStream_t stream) {
    (void)in_sizes; (void)n_in; (void)out_size; (void)ws_size;

    const float* x  = (const float*)d_in[0];  // [1,32,4,128,128]
    const float* W0 = (const float*)d_in[1];  // [256,68,3,3]
    const float* b0 = (const float*)d_in[2];  // [256]
    const float* W1 = (const float*)d_in[3];  // [256,128,3,3]
    const float* b1 = (const float*)d_in[4];  // [256]
    const float* Wr = (const float*)d_in[5];  // [1,64,3,3]
    const float* br = (const float*)d_in[6];  // [1]
    float* y = (float*)d_out;                 // [32,1,128,128]

    char* ws = (char*)d_ws;
    // workspace layout (bytes, all 16B aligned)
    unsigned short* A0p  = (unsigned short*)(ws + 0);          //  256*640*2
    unsigned short* A1p  = (unsigned short*)(ws + 327680);     //  256*1152*2
    float*          c0   = (float*)(ws + 917504);              //  4 MiB
    float*          c1   = (float*)(ws + 5111808);             //  4 MiB
    unsigned short* h0b0 = (unsigned short*)(ws + 9306112);    //  2 MiB
    unsigned short* h1b0 = (unsigned short*)(ws + 11403264);   //  2 MiB
    unsigned short* h0b1 = (unsigned short*)(ws + 13500416);   //  2 MiB
    unsigned short* h1b1 = (unsigned short*)(ws + 15597568);   //  2 MiB
    unsigned short* h0buf[2] = {h0b0, h0b1};
    unsigned short* h1buf[2] = {h1b0, h1b1};

    // zero c0,c1,h0[0],h1[0] (contiguous 12 MiB starting at c0)
    {
        int nfl = (4194304 * 2 + 2097152 * 2) / 4;
        zero_kernel<<<(nfl + 255) / 256, 256, 0, stream>>>(c0, nfl);
    }
    // pack weights (bf16, gate-permuted, K-padded)
    pack_kernel<640, 612><<<(256 * 640) / 256, 256, 0, stream>>>(W0, A0p);
    pack_kernel<1152, 1152><<<(256 * 1152) / 256, 256, 0, stream>>>(W1, A1p);

    for (int t = 0; t < 32; ++t) {
        const unsigned short* h0_rd = h0buf[t & 1];
        unsigned short*       h0_wr = h0buf[(t + 1) & 1];
        const unsigned short* h1_rd = h1buf[t & 1];
        unsigned short*       h1_wr = h1buf[(t + 1) & 1];
        const float* x_t = x + (size_t)t * 4 * 16384;

        cell_kernel<true><<<512, 256, 0, stream>>>(
            x_t, nullptr, h0_rd, A0p, b0, c0, h0_wr);
        cell_kernel<false><<<512, 256, 0, stream>>>(
            nullptr, h0_wr, h1_rd, A1p, b1, c1, h1_wr);
        readout_kernel<<<64, 256, 0, stream>>>(h1_wr, Wr, br, y + (size_t)t * 16384);
    }
}
